// BinaryLinear_19859928776709
// MI455X (gfx1250) — 8-bit matrix operand order repaired and checked by reading, compiled, not run on hardware
//
#include <hip/hip_runtime.h>

typedef __attribute__((ext_vector_type(8))) int   v8i;
typedef __attribute__((ext_vector_type(4))) float f32x4;

__global__ void __launch_bounds__(256)
binarize_f32_to_i8(const float* __restrict__ in, unsigned int* __restrict__ out, int n4) {
  int i = blockIdx.x * blockDim.x + threadIdx.x;
  if (i >= n4) return;
  const f32x4 v = __builtin_nontemporal_load(reinterpret_cast<const f32x4*>(in) + i);
  unsigned int b0 = (v.x > 0.0f) ? 0x01u : 0xFFu;
  unsigned int b1 = (v.y > 0.0f) ? 0x01u : 0xFFu;
  unsigned int b2 = (v.z > 0.0f) ? 0x01u : 0xFFu;
  unsigned int b3 = (v.w > 0.0f) ? 0x01u : 0xFFu;
  out[i] = b0 | (b1 << 8) | (b2 << 16) | (b3 << 24);
}

__global__ void __launch_bounds__(128)
bgemm_wmma_iu8(const char* __restrict__ Aq,
               const char* __restrict__ Bq,
               const float* __restrict__ bias,
               float* __restrict__ out,
               int M, int N, int K)
{
  const int lane = threadIdx.x & 31;
  const int wid  = threadIdx.x >> 5;
  const int r    = lane & 15;
  const int h    = lane >> 4;

  const int wm = wid & 1;
  const int wn = wid >> 1;
  const int m0 = blockIdx.x * 128 + wm * 64;
  const int n0 = blockIdx.y * 64  + wn * 32;

  v8i c[4][2] = {};

  const long kA = (long)h * 16;
  const long kB = (long)h * 16;

  const char* aRow[4];
#pragma unroll
  for (int mt = 0; mt < 4; ++mt) aRow[mt] = Aq + (long)(m0 + mt * 16 + r) * K + kA;
  const char* bRow[2];
#pragma unroll
  for (int nt = 0; nt < 2; ++nt) bRow[nt] = Bq + (long)(n0 + nt * 16 + r) * K + kB;

  for (int k0 = 0; k0 < K; k0 += 64) {
    v8i a[4];
    v8i b[2];

#pragma unroll
    for (int mt = 0; mt < 4; ++mt) {
      const char* p = aRow[mt] + k0;
      const uint4 p0 = *reinterpret_cast<const uint4*>(p);
      const uint4 p1 = *reinterpret_cast<const uint4*>(p + 32);
      a[mt][0] = (int)p0.x; a[mt][1] = (int)p0.y;
      a[mt][2] = (int)p0.z; a[mt][3] = (int)p0.w;
      a[mt][4] = (int)p1.x; a[mt][5] = (int)p1.y;
      a[mt][6] = (int)p1.z; a[mt][7] = (int)p1.w;
      __builtin_prefetch(p + 256, 0, 1);
    }

#pragma unroll
    for (int nt = 0; nt < 2; ++nt) {
      const char* p = bRow[nt] + k0;
      const uint4 q0 = *reinterpret_cast<const uint4*>(p);
      const uint4 q1 = *reinterpret_cast<const uint4*>(p + 32);
      b[nt][0] = (int)q0.x; b[nt][1] = (int)q0.y;
      b[nt][2] = (int)q0.z; b[nt][3] = (int)q0.w;
      b[nt][4] = (int)q1.x; b[nt][5] = (int)q1.y;
      b[nt][6] = (int)q1.z; b[nt][7] = (int)q1.w;
      __builtin_prefetch(p + 256, 0, 1);
    }

#pragma unroll
    for (int mt = 0; mt < 4; ++mt)
#pragma unroll
      for (int nt = 0; nt < 2; ++nt)
        c[mt][nt] = __builtin_amdgcn_wmma_i32_16x16x64_iu8(
            true, a[mt], true, b[nt], c[mt][nt], false, false);
  }

#pragma unroll
  for (int nt = 0; nt < 2; ++nt) {
    const int ocol = n0 + nt * 16 + r;
    const float bv = bias[ocol];
#pragma unroll
    for (int mt = 0; mt < 4; ++mt) {
#pragma unroll
      for (int v = 0; v < 8; ++v) {
        const int orow = m0 + mt * 16 + v + 8 * h;
        __builtin_nontemporal_store((float)c[mt][nt][v] + bv,
                                    out + (long)orow * N + ocol);
      }
    }
  }
}

extern "C" void kernel_launch(void* const* d_in, const int* in_sizes, int n_in,
                              void* d_out, int out_size, void* d_ws, size_t ws_size,
                              hipStream_t stream) {
  const float* x    = (const float*)d_in[0];
  const float* w    = (const float*)d_in[1];
  const float* bias = (const float*)d_in[2];
  float* out        = (float*)d_out;

  const int N = in_sizes[2];
  const int K = in_sizes[1] / N;
  const int M = in_sizes[0] / K;

  char* xq = (char*)d_ws;
  char* wq = xq + (size_t)M * (size_t)K;

  {
    const int n4 = (M * K) / 4;
    binarize_f32_to_i8<<<dim3((n4 + 255) / 256), dim3(256), 0, stream>>>(
        x, (unsigned int*)xq, n4);
  }
  {
    const int n4 = (N * K) / 4;
    binarize_f32_to_i8<<<dim3((n4 + 255) / 256), dim3(256), 0, stream>>>(
        w, (unsigned int*)wq, n4);
  }

  dim3 grid(M / 128, N / 64);
  dim3 block(128);
  bgemm_wmma_iu8<<<grid, block, 0, stream>>>(xq, wq, bias, out, M, N, K);
}
